// LFQ_5162550690365
// MI455X (gfx1250) — matrix operand order repaired and checked by reading, not compiled, not run on hardware
//
#include <hip/hip_runtime.h>
#include <hip/hip_bf16.h>

// ---------------------------------------------------------------------------
// LFQ forward, MI455X (gfx1250, wave32).
//  x:[4,512,2048] f32, W_in:[14,512], b_in:[14], W_out:[512,14], b_out:[512]
//  outputs (concat): out[4,512,2048] f32 | indices[4,2048] i32 | aux_loss f32
//
// codebook = {±1}^14  =>  the 16384-way softmax factorizes into 14 Bernoullis
// p_c = sigmoid(400*h_c).  avg_prob lives as a 64KB LDS histogram filled via
// a 9bit x 5bit outer-product expansion; 8 samples are register-batched so
// each conflict-free wave-wide ds_add_f32 covers 32 bins x 8 samples
// (8x less LDS-atomic traffic than one-sample-at-a-time).
// project_in: v_wmma_f32_16x16x32_f16.  project_out: pure-bandwidth VALU
// (16MB stream write; WMMA would force strided 16x16 stores).
// ---------------------------------------------------------------------------

typedef __attribute__((ext_vector_type(16))) _Float16 v16h;
typedef __attribute__((ext_vector_type(8)))  float    v8f;

#define LFQ_B    4
#define LFQ_D    512
#define LFQ_T    2048
#define LFQ_C    14
#define LFQ_J    16384                    // 2^14 codes
#define LFQ_S    (LFQ_B * LFQ_T)          // 8192 samples
#define OUT_ELEMS (LFQ_B * LFQ_D * LFQ_T) // 4194304
#define IDX_OFF   OUT_ELEMS
#define AUX_OFF   (OUT_ELEMS + LFQ_S)
#define EPSV 1e-20f

// LDS layout (dynamic), kernel 1:
//   accW : float[16384]        65536 B   (avg_prob partial histogram)
//   sW   : _Float16[16*512]    16384 B   (W_in, f16, rows 14/15 zero-padded)
//   hbuf : float[4][16][17]     4352 B   (h, later per-bit probs p_c)
//   sred : float[2]                8 B   (commit, per-sample-entropy partials)
#define SMEM_BYTES (65536 + 16384 + 4352 + 8)
#define HB(w, m, c) ((((w) * 16) + (m)) * 17 + (c))

__global__ void lfq_zero_ws(float* __restrict__ ws) {
    int i = blockIdx.x * blockDim.x + threadIdx.x;
    if (i < LFQ_J + 2) ws[i] = 0.0f;
}

__global__ __attribute__((amdgpu_flat_work_group_size(128, 128)))
void lfq_main(const float* __restrict__ x,
              const float* __restrict__ W_in,
              const float* __restrict__ b_in,
              int*   __restrict__ idx_out,   // -> d_out[IDX_OFF]
              float* __restrict__ ws)        // [16384] hist + [2] scalars
{
    extern __shared__ char smem[];
    float*    accW = (float*)smem;
    _Float16* sW   = (_Float16*)(smem + 65536);
    float*    hbuf = (float*)(smem + 65536 + 16384);
    float*    sred = (float*)(smem + 65536 + 16384 + 4352);

    const int tid  = threadIdx.x;            // 0..127
    const int w    = tid >> 5;               // wave 0..3
    const int lane = tid & 31;
    const int g    = lane >> 4;              // half-wave
    const int hl   = lane & 15;

    // ---- stage: zero histogram, load W_in as f16 (B-matrix friendly) ----
    for (int i = tid; i < LFQ_J; i += 128) accW[i] = 0.0f;
    for (int i = tid; i < 16 * LFQ_D; i += 128)
        sW[i] = (i < LFQ_C * LFQ_D) ? (_Float16)W_in[i] : (_Float16)0.0f;
    if (tid < 2) sred[tid] = 0.0f;
    __syncthreads();

    // ---- GEMM1 per wave: h[16 samples x 16 dims] via WMMA f16 ----
    // A (16x32 f16): lane m=hl holds K = k0 + 16*g + e
    // B (32x16 f16): lane n=hl holds K = k0 + 16*g + e, contiguous in LDS
    const int tileBase = blockIdx.x * 64 + w * 16;
    const int s  = tileBase + hl;
    const int bb = s >> 11;                  // /T
    const int tt = s & (LFQ_T - 1);
    const float* xb = x + (size_t)bb * (LFQ_D * LFQ_T) + tt;

    v8f cfr = {};
    #pragma unroll 2
    for (int k0 = 0; k0 < LFQ_D; k0 += 32) {
        if (k0 + 32 < LFQ_D)
            __builtin_prefetch(xb + (size_t)(k0 + 32 + 8 * g) * LFQ_T, 0, 1); // global_prefetch
        v16h a, bf;
        #pragma unroll
        for (int e = 0; e < 16; ++e) {
            int K = k0 + 16 * g + e;
            a[e] = (_Float16)xb[(size_t)K * LFQ_T];
        }
        const _Float16* bp = sW + hl * LFQ_D + k0 + 16 * g;
        #pragma unroll
        for (int e = 0; e < 16; ++e) bf[e] = bp[e];
        cfr = __builtin_amdgcn_wmma_f32_16x16x32_f16(
                  false, a, false, bf, (short)0, cfr, false, false);
    }
    const float bias = (hl < LFQ_C) ? b_in[hl] : 0.0f;
    #pragma unroll
    for (int r = 0; r < 8; ++r)
        hbuf[HB(w, r + 8 * g, hl)] = cfr[r] + bias;   // h[m][c], C/D layout M=r+8g,N=hl
    __syncthreads();

    // ---- per-sample: sign-quantize, pack index, commit, per-bit entropy ----
    if (tid < 64) {
        const int w2 = tid >> 4, m2 = tid & 15;
        const int s2 = blockIdx.x * 64 + tid;
        int   idx = 0;
        float commit = 0.0f, ent = 0.0f;
        #pragma unroll
        for (int c = 0; c < LFQ_C; ++c) {
            float h   = hbuf[HB(w2, m2, c)];
            int   bit = (h > 0.0f);
            idx |= bit << (13 - c);
            float qv = bit ? 1.0f : -1.0f;
            float dd = h - qv;
            commit += dd * dd;
            float pv = 1.0f / (1.0f + __expf(-400.0f * h));   // factorized softmax
            float q1 = 1.0f - pv;
            ent -= pv * __logf(fmaxf(pv, EPSV))
                 + q1 * __logf(fmaxf(q1, EPSV));
            hbuf[HB(w2, m2, c)] = pv;                         // overwrite h with p_c
        }
        idx_out[s2] = idx;
        atomicAdd(&sred[0], commit);
        atomicAdd(&sred[1], ent);
    }
    __syncthreads();

    // ---- avg_prob histogram: outer-product expansion, 8-sample register batch ----
    // j = [jh:9 bits | lane:5 bits]; bit i of lane -> c = 13-i ; bit i of jh -> c = 8-i
    for (int sg = 0; sg < 2; ++sg) {
        float p[8][LFQ_C], q[8][LFQ_C], fl[8];
        #pragma unroll
        for (int ss = 0; ss < 8; ++ss) {
            #pragma unroll
            for (int c = 0; c < LFQ_C; ++c) {
                float pv = hbuf[HB(w, sg * 8 + ss, c)];       // broadcast ds_load
                p[ss][c] = pv;
                q[ss][c] = 1.0f - pv;
            }
            float flo = 1.0f;
            #pragma unroll
            for (int i = 0; i < 5; ++i)
                flo *= ((lane >> i) & 1) ? p[ss][13 - i] : q[ss][13 - i];
            fl[ss] = flo;
        }
        for (int i2 = 0; i2 < 8; ++i2) {
            float f2[8];
            #pragma unroll
            for (int ss = 0; ss < 8; ++ss)
                f2[ss] = fl[ss] * (((i2 & 1) ? p[ss][2] : q[ss][2])
                                 * ((i2 & 2) ? p[ss][1] : q[ss][1])
                                 * ((i2 & 4) ? p[ss][0] : q[ss][0]));
            for (int i1 = 0; i1 < 8; ++i1) {
                float f21[8];
                #pragma unroll
                for (int ss = 0; ss < 8; ++ss)
                    f21[ss] = f2[ss] * (((i1 & 1) ? p[ss][5] : q[ss][5])
                                      * ((i1 & 2) ? p[ss][4] : q[ss][4])
                                      * ((i1 & 4) ? p[ss][3] : q[ss][3]));
                float* row = accW + ((((i2 << 6) | (i1 << 3)) << 5) + lane);
                #pragma unroll
                for (int i0 = 0; i0 < 8; ++i0) {
                    float sum = 0.0f;
                    #pragma unroll
                    for (int ss = 0; ss < 8; ++ss)
                        sum += f21[ss] * (((i0 & 1) ? p[ss][8] : q[ss][8])
                                        * ((i0 & 2) ? p[ss][7] : q[ss][7])
                                        * ((i0 & 4) ? p[ss][6] : q[ss][6]));
                    atomicAdd(row + (i0 << 5), sum);          // 1 ds_add per 32 bins x 8 samples
                }
            }
        }
    }
    __syncthreads();

    // ---- flush block partials to global ----
    for (int i = tid; i < LFQ_J; i += 128) atomicAdd(&ws[i], accW[i]);
    if (tid == 0) { atomicAdd(&ws[LFQ_J], sred[0]); atomicAdd(&ws[LFQ_J + 1], sred[1]); }
}

// out[b,d,t] = b_out[d] + sum_c (+-1)*W_out[d,c]  — pure bandwidth (16MB write)
__global__ void lfq_project_out(const int* __restrict__ idx_in,
                                const float* __restrict__ W_out,
                                const float* __restrict__ b_out,
                                float* __restrict__ out)
{
    int o = blockIdx.x * 256 + threadIdx.x;          // 0..4194303
    int t    = o & (LFQ_T - 1);
    int rest = o >> 11;
    int d    = rest & (LFQ_D - 1);
    int b    = rest >> 9;
    int idx  = idx_in[b * LFQ_T + t];
    float acc = b_out[d];
    #pragma unroll
    for (int c = 0; c < LFQ_C; ++c) {
        float wv = W_out[d * LFQ_C + c];
        acc += ((idx >> (13 - c)) & 1) ? wv : -wv;
    }
    out[o] = acc;
}

__global__ void lfq_finalize(const float* __restrict__ ws, float* __restrict__ aux)
{
    __shared__ float red[256];
    int tid = threadIdx.x;
    float local = 0.0f;
    for (int i = tid; i < LFQ_J; i += 256) {
        float q = ws[i] * (1.0f / (float)LFQ_S);     // avg_prob
        local -= q * __logf(fmaxf(q, EPSV));
    }
    red[tid] = local;
    __syncthreads();
    for (int stride = 128; stride > 0; stride >>= 1) {
        if (tid < stride) red[tid] += red[tid + stride];
        __syncthreads();
    }
    if (tid == 0) {
        float cb_ent  = red[0];
        float ps_ent  = ws[LFQ_J + 1] * (1.0f / (float)LFQ_S);
        float commit  = ws[LFQ_J]     * (1.0f / (float)(LFQ_S * LFQ_C));
        *aux = (ps_ent - 1.0f * cb_ent) * 0.1f + commit * 1.0f;
    }
}

extern "C" void kernel_launch(void* const* d_in, const int* in_sizes, int n_in,
                              void* d_out, int out_size, void* d_ws, size_t ws_size,
                              hipStream_t stream) {
    const float* x     = (const float*)d_in[0];
    const float* W_in  = (const float*)d_in[1];
    const float* b_in  = (const float*)d_in[2];
    const float* W_out = (const float*)d_in[3];
    const float* b_out = (const float*)d_in[4];

    float* out_f  = (float*)d_out;
    int*   idx_p  = (int*)(out_f + IDX_OFF);
    float* aux_p  = out_f + AUX_OFF;
    float* ws     = (float*)d_ws;   // [16384] histogram + [2] scalars

    lfq_zero_ws<<<(LFQ_J + 2 + 255) / 256, 256, 0, stream>>>(ws);
    lfq_main<<<LFQ_S / 64, 128, SMEM_BYTES, stream>>>(x, W_in, b_in, idx_p, ws);
    lfq_project_out<<<OUT_ELEMS / 256, 256, 0, stream>>>(idx_p, W_out, b_out, out_f);
    lfq_finalize<<<1, 256, 0, stream>>>(ws, aux_p);
}
